// LSTMForecaster_51634096833233
// MI455X (gfx1250) — matrix operand order repaired and checked by reading, not compiled, not run on hardware
//
#include <hip/hip_runtime.h>
#include <hip/hip_bf16.h>

// ---------------------------------------------------------------------------
// Fused seq2seq LSTM for MI455X (gfx1250, wave32, WMMA 16x16x32 f16).
//
// Strategy:
//  * One persistent kernel: each 256-thread block (8 waves) owns 64 batch rows
//    and runs the whole encoder(20 steps x 2 layers) + decoder(30 x 2) + head.
//  * Wave u owns hidden units [16u,16u+16) of ALL 4 gates -> cell state c
//    lives in that wave's VGPRs across all 100 steps.
//  * Weights are pre-packed (f16) into exact WMMA B-fragment lane layout.
//    Recurrent Whh matrices (256KB) are LDS-resident per phase; Wih streams
//    from L2 (192MB -> always resident).
//  * h kept in LDS (f16, row stride 136 halfs -> conflict-free b128 A-frag
//    loads); new h stashed in regs, written after a barrier (single buffer).
//
// WMMA fragment layout assumed (from CDNA5 ISA 7.12.2, 16-bit A 16x32):
//   lane l: row/col = l&15, khalf = l>>4
//   element j -> K = j + (j>=8 ? 8 : 0) + khalf*8   (A by rows, B by cols)
//   D/C f32: element r -> row = r + (l>>4)*8, col = l&15
// ---------------------------------------------------------------------------

typedef __attribute__((ext_vector_type(16))) _Float16 v16h;
typedef __attribute__((ext_vector_type(8)))  float    v8f;

union FragU { v16h v; uint4 q[2]; };

// packed-weight offsets in d_ws (units: halfs). Per matrix: 32 ntiles * KT
// ktiles * 32 lanes * 16 halfs.
#define WOFF_EWIH0 0
#define WOFF_EWHH0 16384
#define WOFF_EWIH1 81920
#define WOFF_EWHH1 147456
#define WOFF_DWIH0 212992
#define WOFF_DWHH0 229376
#define WOFF_DWIH1 294912
#define WOFF_DWHH1 360448
#define HALF_TOTAL 425984   // bytes*2 = 851968; combined biases (f32) follow

__device__ __forceinline__ v8f wmma16(v16h a, v16h b, v8f c) {
  return __builtin_amdgcn_wmma_f32_16x16x32_f16(
      /*neg_a=*/false, a, /*neg_b=*/false, b,
      /*c_mod=*/(short)0, c, /*reuse_a=*/false, /*reuse_b=*/false);
}

// A fragment: 16 rows (srow..srow+15) x 32 K from an f16 row-major LDS buffer.
__device__ __forceinline__ v16h load_afrag(const _Float16* buf, int stride,
                                           int srow, int kt, int lane) {
  int m  = srow + (lane & 15);
  int kh = (lane >> 4) << 3;                 // 0 or 8
  const _Float16* p = buf + m * stride + kt * 32 + kh;
  FragU f;
  f.q[0] = *(const uint4*)(p);               // K = kh .. kh+7
  f.q[1] = *(const uint4*)(p + 16);          // K = kh+16 .. kh+23
  return f.v;
}

// B fragment from pre-packed weights (global/L2 or LDS copy, identical layout)
__device__ __forceinline__ v16h load_bfrag(const _Float16* wpack, int j,
                                           int KT, int k, int lane) {
  const _Float16* p = wpack + (((j * KT) + k) * 32 + lane) * 16;
  FragU f;
  f.q[0] = *(const uint4*)(p);
  f.q[1] = *(const uint4*)(p + 8);
  return f.v;
}

__device__ __forceinline__ float sigf(float x) {
  return 1.0f / (1.0f + __expf(-x));
}
__device__ __forceinline__ float tanhfast(float x) {
  float e = __expf(2.0f * x);
  return 1.0f - 2.0f / (e + 1.0f);           // safe at +-inf
}

// One LSTM sub-layer for this wave: gates for hidden tile u over 64 rows.
// Reads xsrc (K = 32*KTX) and hsrc (K = 128) from LDS, weights packed.
// Updates cst[32] (VGPR), produces hst[32] (new h, f16) WITHOUT writing LDS.
template <int KTX>
__device__ __forceinline__ void lstm_sub(const _Float16* xsrc, int xstride,
                                         const _Float16* hsrc,
                                         const _Float16* wx,  // packed, KT=KTX
                                         const _Float16* wh,  // packed, KT=4
                                         float4 bias, float* cst,
                                         _Float16* hst, int lane, int u) {
#pragma unroll
  for (int s = 0; s < 4; ++s) {              // 4 x 16-row subtiles
    v16h ax[KTX];
    v16h ah[4];
#pragma unroll
    for (int k = 0; k < KTX; ++k) ax[k] = load_afrag(xsrc, xstride, s * 16, k, lane);
#pragma unroll
    for (int k = 0; k < 4; ++k)   ah[k] = load_afrag(hsrc, 136, s * 16, k, lane);

    v8f acc[4];
#pragma unroll
    for (int g = 0; g < 4; ++g) {            // gates i,f,g,o -> ntile j=8g+u
      int j = g * 8 + u;
      v8f a = {0.f, 0.f, 0.f, 0.f, 0.f, 0.f, 0.f, 0.f};
#pragma unroll
      for (int k = 0; k < KTX; ++k) a = wmma16(ax[k], load_bfrag(wx, j, KTX, k, lane), a);
#pragma unroll
      for (int k = 0; k < 4; ++k)   a = wmma16(ah[k], load_bfrag(wh, j, 4, k, lane), a);
      acc[g] = a;
    }
#pragma unroll
    for (int r = 0; r < 8; ++r) {
      float gi = sigf(acc[0][r] + bias.x);
      float gf = sigf(acc[1][r] + bias.y);
      float gg = tanhfast(acc[2][r] + bias.z);
      float go = sigf(acc[3][r] + bias.w);
      float c  = gf * cst[s * 8 + r] + gi * gg;
      cst[s * 8 + r] = c;
      hst[s * 8 + r] = (_Float16)(go * tanhfast(c));
    }
  }
}

// Commit stashed h (after a block barrier). D-frag mapping: row = s*16+r+khalf*8
__device__ __forceinline__ void write_h(_Float16* hdst, const _Float16* hst,
                                        int lane, int u) {
  int colb = u * 16 + (lane & 15);
  int rb   = (lane >> 4) << 3;
#pragma unroll
  for (int s = 0; s < 4; ++s)
#pragma unroll
    for (int r = 0; r < 8; ++r)
      hdst[(s * 16 + r + rb) * 136 + colb] = hst[s * 8 + r];
}

// ---------------------------------------------------------------------------
// Pre-pack kernels
// ---------------------------------------------------------------------------

// Pack W (512 x Ksrc, f32, PyTorch layout) into WMMA B-fragment f16 layout,
// zero-padding K up to 32*KT. dst linear idx = ((j*KT+k)*32+lane)*16 + jh.
__global__ void pack_weights_k(const float* __restrict__ W,
                               _Float16* __restrict__ dst, int Ksrc, int KT) {
  int idx = blockIdx.x * 256 + threadIdx.x;
  int total = KT * 16384;                    // 32*KT*512
  if (idx >= total) return;
  int jh    = idx & 15;
  int lane  = (idx >> 4) & 31;
  int rem   = idx >> 9;                      // j*KT + k
  int k     = rem % KT;
  int j     = rem / KT;
  int nn    = lane & 15;
  int khalf = lane >> 4;
  int kk    = 32 * k + jh + ((jh >= 8) ? 8 : 0) + khalf * 8;
  int rowW  = 16 * j + nn;                   // gate/output column 0..511
  float v = (kk < Ksrc) ? W[rowW * Ksrc + kk] : 0.0f;
  dst[idx] = (_Float16)v;
}

__global__ void pack_bias_k(const float* __restrict__ a,
                            const float* __restrict__ b,
                            float* __restrict__ dst) {
  int i = blockIdx.x * 256 + threadIdx.x;
  if (i < 512) dst[i] = a[i] + b[i];
}

// ---------------------------------------------------------------------------
// Main fused kernel
// ---------------------------------------------------------------------------
__global__ __launch_bounds__(256, 1) void lstm_forecast(
    const float* __restrict__ hist, const _Float16* __restrict__ wp,
    const float* __restrict__ biasC, const float* __restrict__ headW,
    const float* __restrict__ headB, float* __restrict__ out,
    const int* __restrict__ futlen) {
  // LDS: 2 resident Whh (256KB) + h0/h1 (2x17KB) + x tile (5KB) = ~295KB
  __shared__ __align__(16) _Float16 wlds[131072];   // [2][32][4][32][16]
  __shared__ __align__(16) _Float16 hb0[8704];      // 64 x 136
  __shared__ __align__(16) _Float16 hb1[8704];
  __shared__ __align__(16) _Float16 xb[2560];       // 64 x 40 (K padded to 32)

  const int tid  = threadIdx.x;
  const int lane = tid & 31;
  const int u    = tid >> 5;                 // wave id == hidden tile id
  const int TFUT = futlen[0];
  const int row0 = blockIdx.x * 64;

  for (int i = tid; i < 8704; i += 256) { hb0[i] = (_Float16)0.f; hb1[i] = (_Float16)0.f; }
  {  // stage encoder Whh matrices into LDS
    const uint4* s0 = (const uint4*)(wp + WOFF_EWHH0);
    const uint4* s1 = (const uint4*)(wp + WOFF_EWHH1);
    uint4* d0 = (uint4*)(wlds);
    uint4* d1 = (uint4*)(wlds + 65536);
    for (int i = tid; i < 8192; i += 256) { d0[i] = s0[i]; d1[i] = s1[i]; }
  }
  const int bi = u * 16 + (lane & 15);       // this lane's hidden-unit column
  float4 bE0 = make_float4(biasC[bi],        biasC[128 + bi],  biasC[256 + bi],  biasC[384 + bi]);
  float4 bE1 = make_float4(biasC[512 + bi],  biasC[640 + bi],  biasC[768 + bi],  biasC[896 + bi]);
  float4 bD0 = make_float4(biasC[1024 + bi], biasC[1152 + bi], biasC[1280 + bi], biasC[1408 + bi]);
  float4 bD1 = make_float4(biasC[1536 + bi], biasC[1664 + bi], biasC[1792 + bi], biasC[1920 + bi]);

  float c0[32], c1[32];
#pragma unroll
  for (int i = 0; i < 32; ++i) { c0[i] = 0.f; c1[i] = 0.f; }
  _Float16 hst[32];
  __syncthreads();

  // ------------------------------ encoder ------------------------------
#pragma unroll 1
  for (int t = 0; t < 20; ++t) {
    for (int i = tid; i < 2048; i += 256) {  // stage x_t (F=10, pad to 32)
      int r = i >> 5, c = i & 31;
      float v = (c < 10) ? hist[(size_t)(row0 + r) * 200 + t * 10 + c] : 0.0f;
      xb[r * 40 + c] = (_Float16)v;
    }
    __syncthreads();
    lstm_sub<1>(xb, 40, hb0, wp + WOFF_EWIH0, wlds, bE0, c0, hst, lane, u);
    __syncthreads();
    write_h(hb0, hst, lane, u);
    __syncthreads();
    lstm_sub<4>(hb0, 136, hb1, wp + WOFF_EWIH1, wlds + 65536, bE1, c1, hst, lane, u);
    __syncthreads();
    write_h(hb1, hst, lane, u);
    __syncthreads();
  }

  // ------------------- swap resident weights to decoder ----------------
  {
    const uint4* s0 = (const uint4*)(wp + WOFF_DWHH0);
    const uint4* s1 = (const uint4*)(wp + WOFF_DWHH1);
    uint4* d0 = (uint4*)(wlds);
    uint4* d1 = (uint4*)(wlds + 65536);
    for (int i = tid; i < 8192; i += 256) { d0[i] = s0[i]; d1[i] = s1[i]; }
  }
  for (int i = tid; i < 2560; i += 256) xb[i] = (_Float16)0.f;  // y0 = 0
  __syncthreads();

  // ------------------------------ decoder ------------------------------
#pragma unroll 1
  for (int ft = 0; ft < TFUT; ++ft) {
    lstm_sub<1>(xb, 40, hb0, wp + WOFF_DWIH0, wlds, bD0, c0, hst, lane, u);
    __syncthreads();
    write_h(hb0, hst, lane, u);
    __syncthreads();
    lstm_sub<4>(hb0, 136, hb1, wp + WOFF_DWIH1, wlds + 65536, bD1, c1, hst, lane, u);
    __syncthreads();
    write_h(hb1, hst, lane, u);
    __syncthreads();
    if (tid < 128) {                          // head: y = h1 @ headW^T + b
      int r = tid >> 1, oc = tid & 1;
      float acc = headB[oc];
#pragma unroll 16
      for (int k = 0; k < 128; ++k)
        acc += (float)hb1[r * 136 + k] * headW[oc * 128 + k];
      out[(size_t)(row0 + r) * (TFUT * 2) + ft * 2 + oc] = acc;
      xb[r * 40 + oc] = (_Float16)acc;        // next-step decoder input
    }
    __syncthreads();
  }
}

// ---------------------------------------------------------------------------
extern "C" void kernel_launch(void* const* d_in, const int* in_sizes, int n_in,
                              void* d_out, int out_size, void* d_ws, size_t ws_size,
                              hipStream_t stream) {
  const float* hist  = (const float*)d_in[0];
  const float* eWih0 = (const float*)d_in[1];
  const float* eWhh0 = (const float*)d_in[2];
  const float* eb0a  = (const float*)d_in[3];
  const float* eb0b  = (const float*)d_in[4];
  const float* eWih1 = (const float*)d_in[5];
  const float* eWhh1 = (const float*)d_in[6];
  const float* eb1a  = (const float*)d_in[7];
  const float* eb1b  = (const float*)d_in[8];
  const float* dWih0 = (const float*)d_in[9];
  const float* dWhh0 = (const float*)d_in[10];
  const float* db0a  = (const float*)d_in[11];
  const float* db0b  = (const float*)d_in[12];
  const float* dWih1 = (const float*)d_in[13];
  const float* dWhh1 = (const float*)d_in[14];
  const float* db1a  = (const float*)d_in[15];
  const float* db1b  = (const float*)d_in[16];
  const float* headW = (const float*)d_in[17];
  const float* headB = (const float*)d_in[18];
  const int*   futp  = (const int*)d_in[19];

  _Float16* wp = (_Float16*)d_ws;
  float* biasC = (float*)((char*)d_ws + (size_t)HALF_TOTAL * 2);

  auto pk = [&](const float* W, int off, int Ksrc, int KT) {
    int total = KT * 16384;
    pack_weights_k<<<(total + 255) / 256, 256, 0, stream>>>(W, wp + off, Ksrc, KT);
  };
  pk(eWih0, WOFF_EWIH0, 10, 1);
  pk(eWhh0, WOFF_EWHH0, 128, 4);
  pk(eWih1, WOFF_EWIH1, 128, 4);
  pk(eWhh1, WOFF_EWHH1, 128, 4);
  pk(dWih0, WOFF_DWIH0, 2, 1);
  pk(dWhh0, WOFF_DWHH0, 128, 4);
  pk(dWih1, WOFF_DWIH1, 128, 4);
  pk(dWhh1, WOFF_DWHH1, 128, 4);
  pack_bias_k<<<2, 256, 0, stream>>>(eb0a, eb0b, biasC + 0);
  pack_bias_k<<<2, 256, 0, stream>>>(eb1a, eb1b, biasC + 512);
  pack_bias_k<<<2, 256, 0, stream>>>(db0a, db0b, biasC + 1024);
  pack_bias_k<<<2, 256, 0, stream>>>(db1a, db1b, biasC + 1536);

  int B = in_sizes[0] / 200;                 // hist = (B, 20, 10)
  lstm_forecast<<<B / 64, 256, 0, stream>>>(hist, wp, biasC, headW, headB,
                                            (float*)d_out, futp);
}
